// MLSACoreAttention_25812753449396
// MI455X (gfx1250) — matrix operand order repaired and checked by reading, not compiled, not run on hardware
//
#include <hip/hip_runtime.h>

// ---------------------------------------------------------------------------
// MLSA core attention (causal, no-max-sub softmax) for MI455X / gfx1250.
// Flash-attention structure on v_wmma_f32_16x16x32_f16 (wave32 WMMA).
// ---------------------------------------------------------------------------

typedef __attribute__((ext_vector_type(16))) _Float16 v16h;
typedef __attribute__((ext_vector_type(8)))  _Float16 v8h;
typedef __attribute__((ext_vector_type(8)))  float    v8f;
typedef __attribute__((ext_vector_type(4)))  float    v4f;

#define SQt     2048
#define SKt     2048
#define BATCH   2
#define NHEADS  16
#define HN      128
#define QB      128   // query rows per workgroup
#define KT      32    // key-tile (one WMMA K-sweep of 32 keys)
#define NW      8     // waves per workgroup (wave32)

// total scale = SCALE / NORM_FACTOR = 1/sqrt(128)
#define SCL 0.08838834764831845f

// K tile in LDS: [32 keys][128 + 8 pad] f16 (row stride 136h = 68 dwords -> bank-spread)
#define KLSTRIDE 136
// V^T tile in LDS: [128 h'][32 + 8 pad] f16 (row stride 40h = 20 dwords -> bank-spread)
#define VTSTRIDE 40

__device__ __forceinline__ v16h cat8(v8h lo, v8h hi) {
    return __builtin_shufflevector(lo, hi, 0,1,2,3,4,5,6,7,8,9,10,11,12,13,14,15);
}

__global__ __launch_bounds__(256)
void mlsa_attn_kernel(const float* __restrict__ Q,
                      const float* __restrict__ K,
                      const float* __restrict__ V,
                      float* __restrict__ ctx_out,
                      float* __restrict__ den_out)
{
    __shared__ __align__(16) _Float16 Kl[2][KT][KLSTRIDE];
    __shared__ __align__(16) _Float16 Vt[2][HN][VTSTRIDE];
    __shared__ __align__(16) _Float16 Pst[NW][16][32];     // per-wave P staging

    const int tid  = threadIdx.x;
    const int w    = tid >> 5;      // wave id (wave32)
    const int lane = tid & 31;
    const int half = lane >> 4;
    const int ln16 = lane & 15;

    const int q0 = blockIdx.x * QB;
    const int b  = blockIdx.y / NHEADS;
    const int n  = blockIdx.y % NHEADS;

    const int row_stride = BATCH * NHEADS * HN;   // stride between seq positions (floats)
    const int head_off   = (b * NHEADS + n) * HN; // offset of (b,n) inside one seq position

    // ---------------- Q A-fragments (this wave's 16 rows, all 128 dims) ------
    // A layout (16b, 16x32): lane l -> row M = l%16; elements 0..7  <-> k = 8*half + {0..7}
    //                                            elements 8..15 <-> k = 16 + 8*half + {0..7}
    const int   qrow_a = q0 + 16 * w + ln16;
    const float* qptr  = Q + (size_t)qrow_a * row_stride + head_off;
    v16h aq[4];
#pragma unroll
    for (int kc = 0; kc < 4; ++kc) {
        const float* p0 = qptr + 32 * kc + 8 * half;
        const float* p1 = p0 + 16;
        v4f a0 = *(const v4f*)p0,       a1 = *(const v4f*)(p0 + 4);
        v4f b0 = *(const v4f*)p1,       b1 = *(const v4f*)(p1 + 4);
        v16h x;
#pragma unroll
        for (int e = 0; e < 4; ++e) {
            x[e]      = (_Float16)a0[e];
            x[4 + e]  = (_Float16)a1[e];
            x[8 + e]  = (_Float16)b0[e];
            x[12 + e] = (_Float16)b1[e];
        }
        aq[kc] = x;
    }

    // ---------------- persistent state ---------------------------------------
    v8f acc[8];                      // ctx accumulators: acc[c][r] = (row r+8*half, h'=16c+ln16)
#pragma unroll
    for (int c = 0; c < 8; ++c) acc[c] = (v8f){0.f,0.f,0.f,0.f,0.f,0.f,0.f,0.f};
    float rowsum[8] = {0.f,0.f,0.f,0.f,0.f,0.f,0.f,0.f};

    // ---------------- cooperative KV tile staging ----------------------------
    const int crow = tid >> 3;       // 0..31 : key row within tile
    const int ccg  = tid & 7;        // 0..7  : 16-wide column group
    v4f kr[4], vr[4];                // pipeline registers (next tile in flight)

    auto glob_load = [&](int tile) {
        const int t = tile * KT + crow;
        const float* kp = K + (size_t)t * row_stride + head_off + 16 * ccg;
        const float* vp = V + (size_t)t * row_stride + head_off + 16 * ccg;
        kr[0] = *(const v4f*)(kp);      kr[1] = *(const v4f*)(kp + 4);
        kr[2] = *(const v4f*)(kp + 8);  kr[3] = *(const v4f*)(kp + 12);
        vr[0] = *(const v4f*)(vp);      vr[1] = *(const v4f*)(vp + 4);
        vr[2] = *(const v4f*)(vp + 8);  vr[3] = *(const v4f*)(vp + 12);
    };
    auto store_lds = [&](int buf) {
        v8h h0, h1;
#pragma unroll
        for (int e = 0; e < 4; ++e) {
            h0[e]     = (_Float16)kr[0][e];  h0[4 + e] = (_Float16)kr[1][e];
            h1[e]     = (_Float16)kr[2][e];  h1[4 + e] = (_Float16)kr[3][e];
        }
        *(v8h*)&Kl[buf][crow][16 * ccg]     = h0;
        *(v8h*)&Kl[buf][crow][16 * ccg + 8] = h1;
#pragma unroll
        for (int j = 0; j < 16; ++j)
            Vt[buf][16 * ccg + j][crow] = (_Float16)vr[j >> 2][j & 3];
    };

    const int nt = (q0 + QB) / KT;   // causal: only tiles up to the block diagonal
    const int wave_max_row = q0 + 16 * w + 15;

    glob_load(0);
    store_lds(0);
    __syncthreads();

    for (int i = 0; i < nt; ++i) {
        const int p = i & 1;
        const bool have_next = (i + 1) < nt;
        if (have_next) glob_load(i + 1);          // overlap global latency w/ WMMA

        const int t0 = i * KT;
        if (t0 <= wave_max_row) {
            // ---------------- S = Q K^T (two 16x16 D-tiles, 4 k-chunks each) --
#pragma unroll
            for (int sub = 0; sub < 2; ++sub) {
                v8f s = (v8f){0.f,0.f,0.f,0.f,0.f,0.f,0.f,0.f};
#pragma unroll
                for (int kc = 0; kc < 4; ++kc) {
                    // B layout (16b, 32x16): lane l -> col N=l%16; elem e<8 <-> k = 8*half + e, e>=8 <-> k = 16 + 8*half + (e-8)
                    const v8h* kp = (const v8h*)&Kl[p][16 * sub + ln16][32 * kc + 8 * half];
                    v16h bk = cat8(kp[0], kp[2]);
                    s = __builtin_amdgcn_wmma_f32_16x16x32_f16(
                            false, aq[kc], false, bk, (short)0, s, false, false);
                }
                // -------- scale + causal mask + exp (no max-subtraction) ------
                const int key = t0 + 16 * sub + ln16;
#pragma unroll
                for (int r = 0; r < 8; ++r) {
                    const int qr = q0 + 16 * w + r + 8 * half;  // D: M = r + 8*half
                    float e = (key <= qr) ? __expf(s[r] * SCL) : 0.f;  // exp(-10000)->0
                    rowsum[r] += e;
                    Pst[w][r + 8 * half][16 * sub + ln16] = (_Float16)e;
                }
            }
            // ---------------- P A-fragment (D-layout -> A-layout via LDS) ----
            const v8h* pp0 = (const v8h*)&Pst[w][ln16][8 * half];
            const v8h* pp1 = (const v8h*)&Pst[w][ln16][16 + 8 * half];
            v16h ap = cat8(pp0[0], pp1[0]);
            // ---------------- ctx += P V (8 column chunks of 16) -------------
#pragma unroll
            for (int c = 0; c < 8; ++c) {
                const v8h* vp = (const v8h*)&Vt[p][16 * c + ln16][8 * half];
                v16h bv = cat8(vp[0], vp[2]);
                acc[c] = __builtin_amdgcn_wmma_f32_16x16x32_f16(
                             false, ap, false, bv, (short)0, acc[c], false, false);
            }
        }
        __syncthreads();                          // everyone done reading buf p
        if (have_next) {
            store_lds(1 - p);                     // safe: last read of 1-p synced at i-1
            __syncthreads();
        }
    }

    // ---------------- epilogue: denom reduce, scale, store --------------------
#pragma unroll
    for (int r = 0; r < 8; ++r) {
        float s = rowsum[r];
        s += __shfl_xor(s, 1, 32);
        s += __shfl_xor(s, 2, 32);
        s += __shfl_xor(s, 4, 32);
        s += __shfl_xor(s, 8, 32);                // full 16-lane group sum
        const int qr = q0 + 16 * w + r + 8 * half;
        if (ln16 == 0)
            den_out[((size_t)b * NHEADS + n) * SQt + qr] = s;
        const float inv = 1.f / s;
        float* orow = ctx_out + (size_t)qr * row_stride + (size_t)n * HN + ln16
                      + (size_t)b * (NHEADS * HN);
#pragma unroll
        for (int c = 0; c < 8; ++c)
            orow[16 * c] = acc[c][r] * inv;
    }
}

extern "C" void kernel_launch(void* const* d_in, const int* in_sizes, int n_in,
                              void* d_out, int out_size, void* d_ws, size_t ws_size,
                              hipStream_t stream) {
    (void)in_sizes; (void)n_in; (void)out_size; (void)d_ws; (void)ws_size;
    const float* Q = (const float*)d_in[0];
    const float* K = (const float*)d_in[1];
    const float* V = (const float*)d_in[2];
    // d_in[3] is the causal mask; causality is applied analytically in-kernel.
    float* ctx = (float*)d_out;
    float* den = (float*)d_out + (size_t)SQt * BATCH * NHEADS * HN;

    dim3 grid(SQt / QB, BATCH * NHEADS);
    mlsa_attn_kernel<<<grid, 256, 0, stream>>>(Q, K, V, ctx, den);
}
